// NoisyTransformerEncoderBlock_65360812311337
// MI455X (gfx1250) — matrix operand order repaired and checked by reading, not compiled, not run on hardware
//
#include <hip/hip_runtime.h>
#include <math.h>

typedef __attribute__((ext_vector_type(16))) __bf16 v16bf;
typedef __attribute__((ext_vector_type(8)))  float  v8f;
typedef __attribute__((ext_vector_type(2)))  __bf16 bf16x2;
typedef __attribute__((ext_vector_type(2)))  float  f32x2;
typedef __attribute__((ext_vector_type(2)))  unsigned u32x2;
typedef __attribute__((ext_vector_type(4)))  unsigned u32x4;

#define kD 1024
#define kS 2048
#define kB 32
#define kH 16
#define SSUB 32          // sequence rows per sub-tile (== WMMA K for GEMM2)
#define NCHUNK 8         // s-chunks per batch (grid.y)
#define ROWS_PER_WG (kS / NCHUNK)        // 256
#define SUBS_PER_WG (ROWS_PER_WG / SSUB) // 8

// ---------- helpers ----------
static __device__ __forceinline__ unsigned pack2(float x, float y) {
  f32x2 f; f[0] = x; f[1] = y;
  bf16x2 b = __builtin_convertvector(f, bf16x2);  // v_cvt_pk_bf16_f32
  union { bf16x2 b; unsigned u; } o; o.b = b;
  return o.u;
}

union Frag { v16bf v; u32x4 q[2]; };

// 16 consecutive bf16 (32B, 16B-aligned) -> fragment elements 0..15
static __device__ __forceinline__ v16bf load_frag_contig(const __bf16* p) {
  Frag f;
  f.q[0] = *(const u32x4*)(p);
  f.q[1] = *(const u32x4*)(p + 8);
  return f.v;
}
// two 8-element runs (A-matrix 'amap' layout) -> elements 0..7, 8..15
static __device__ __forceinline__ v16bf load_frag_two(const __bf16* p0, const __bf16* p1) {
  Frag f;
  f.q[0] = *(const u32x4*)(p0);
  f.q[1] = *(const u32x4*)(p1);
  return f.v;
}

// =====================================================================
// Kernel A: qtilde[b][h][:] = (q[b,h,:] @ Wk_h) / 8,  q = z_prev @ Wq^T + bq
// =====================================================================
__global__ void __launch_bounds__(256) qtilde_kernel(
    const float* __restrict__ z_prev, const float* __restrict__ in_proj_w,
    const float* __restrict__ in_proj_b, float* __restrict__ qtilde) {
  __shared__ float zl[kD];
  __shared__ float ql[64];
  const int b = blockIdx.x, h = blockIdx.y, tid = threadIdx.x;
  for (int i = tid; i < kD; i += 256) zl[i] = z_prev[b * kD + i];
  __syncthreads();
  if (tid < 64) {
    const float* wr = in_proj_w + (size_t)(h * 64 + tid) * kD;  // Wq rows
    float s = in_proj_b[h * 64 + tid];
    for (int e = 0; e < kD; ++e) s += wr[e] * zl[e];
    ql[tid] = s;
  }
  __syncthreads();
  for (int col = tid; col < kD; col += 256) {
    const float* wk = in_proj_w + (size_t)(kD + h * 64) * kD + col;  // Wk rows
    float s = 0.f;
    for (int d = 0; d < 64; ++d) s += ql[d] * wk[(size_t)d * kD];
    qtilde[((size_t)b * kH + h) * kD + col] = s * 0.125f;  // 1/sqrt(64)
  }
}

// =====================================================================
// Kernel B: streaming attention over x_feat with online softmax.
//   xs[s][d] feeds GEMM1 B-frags (contract over D) with 2x ds_load_b128
//   xt[d][s] feeds GEMM2 B-frags (contract over s') with 2x ds_load_b128
// =====================================================================
__global__ void __launch_bounds__(256) attn_stream_kernel(
    const float* __restrict__ x_feat, const float* __restrict__ qtilde,
    float* __restrict__ pm, float* __restrict__ pl, float* __restrict__ pacc) {
  extern __shared__ char smem[];
  __bf16* xs     = (__bf16*)smem;                         // [32][1024]  65536 B
  __bf16* xt     = (__bf16*)(smem + 65536);               // [1024][32]  65536 B
  __bf16* pbuf   = (__bf16*)(smem + 131072);              // [16][32]     1024 B
  float*  scores = (float*)(smem + 132096);               // [16][32]     2048 B
  float*  mrow   = (float*)(smem + 134144);               // 16
  float*  lrow   = mrow + 16;
  float*  rs     = lrow + 16;

  const int b = blockIdx.x;
  const int cchunk = blockIdx.y;
  const int tid = threadIdx.x;
  const int wave = tid >> 5;
  const int lane = tid & 31;
  const int hi = lane >> 4;
  const int ln = lane & 15;

  if (tid < 16) { mrow[tid] = -1e30f; lrow[tid] = 0.f; }

  // Persistent qtilde A-fragments for this wave's K-range (wave*128 .. +127)
  v16bf aq[4];
  {
    const float* qt = qtilde + (size_t)b * kH * kD + (size_t)ln * kD;
#pragma unroll
    for (int kk = 0; kk < 4; ++kk) {
      int k0 = wave * 128 + kk * 32;
      v16bf a;
#pragma unroll
      for (int j = 0; j < 8; ++j) a[j] = (__bf16)qt[k0 + hi * 8 + j];
#pragma unroll
      for (int j = 0; j < 8; ++j) a[8 + j] = (__bf16)qt[k0 + 16 + hi * 8 + j];
      aq[kk] = a;
    }
  }

  v8f acc[8];
#pragma unroll
  for (int t = 0; t < 8; ++t)
#pragma unroll
    for (int r = 0; r < 8; ++r) acc[t][r] = 0.f;

  const float* xb = x_feat + ((size_t)b * kS + (size_t)cchunk * ROWS_PER_WG) * kD;

  for (int sc = 0; sc < SUBS_PER_WG; ++sc) {
    __syncthreads();  // xs/xt reuse from previous iteration
    const float* xsrc = xb + (size_t)sc * SSUB * kD;
    // stage x sub-tile as bf16 into both layouts
    for (int i = tid; i < (SSUB * kD) / 4; i += 256) {
      float4 f = ((const float4*)xsrc)[i];
      u32x2 pk; pk[0] = pack2(f.x, f.y); pk[1] = pack2(f.z, f.w);
      ((u32x2*)xs)[i] = pk;                    // ds_store_b64
      int s = (i * 4) >> 10;
      int d = (i * 4) & (kD - 1);
      xt[(d + 0) * SSUB + s] = (__bf16)f.x;
      xt[(d + 1) * SSUB + s] = (__bf16)f.y;
      xt[(d + 2) * SSUB + s] = (__bf16)f.z;
      xt[(d + 3) * SSUB + s] = (__bf16)f.w;
    }
    for (int i = tid; i < kH * SSUB; i += 256) scores[i] = 0.f;
    if (sc + 1 < SUBS_PER_WG)  // gfx1250 global_prefetch_b8 for next tile
      __builtin_prefetch(xsrc + SSUB * kD + (size_t)tid * 128, 0, 0);
    __syncthreads();

    // ---- GEMM1: partial scores over this wave's K-range, both N-tiles ----
#pragma unroll
    for (int nt = 0; nt < 2; ++nt) {
      v8f sa;
#pragma unroll
      for (int r = 0; r < 8; ++r) sa[r] = 0.f;
#pragma unroll
      for (int kk = 0; kk < 4; ++kk) {
        int k0 = wave * 128 + kk * 32;
        v16bf bf = load_frag_two(xs + (nt * 16 + ln) * kD + k0 + hi * 8,
                                 xs + (nt * 16 + ln) * kD + k0 + 16 + hi * 8);
        sa = __builtin_amdgcn_wmma_f32_16x16x32_bf16(
            false, aq[kk], false, bf, (short)0, sa, false, false);
      }
#pragma unroll
      for (int r = 0; r < 8; ++r)
        atomicAdd(&scores[(r + hi * 8) * SSUB + nt * 16 + ln], sa[r]);
    }
    __syncthreads();

    // ---- chunk softmax (one thread per head) ----
    if (tid < kH) {
      const int h = tid;
      float mOld = mrow[h], cmax = -1e30f;
      for (int j = 0; j < SSUB; ++j) cmax = fmaxf(cmax, scores[h * SSUB + j]);
      float mNew = fmaxf(mOld, cmax);
      float resc = __expf(mOld - mNew);
      float ls = 0.f;
      for (int j = 0; j < SSUB; ++j) {
        float e = __expf(scores[h * SSUB + j] - mNew);
        ls += e;
        pbuf[h * SSUB + j] = (__bf16)e;
      }
      lrow[h] = lrow[h] * resc + ls;
      mrow[h] = mNew;
      rs[h] = resc;
    }
    __syncthreads();

    // ---- rescale accumulators, then GEMM2 ----
    float4 f0 = *(const float4*)(rs + hi * 8);
    float4 f1 = *(const float4*)(rs + hi * 8 + 4);
    float fsc[8] = {f0.x, f0.y, f0.z, f0.w, f1.x, f1.y, f1.z, f1.w};
#pragma unroll
    for (int t = 0; t < 8; ++t)
#pragma unroll
      for (int r = 0; r < 8; ++r) acc[t][r] *= fsc[r];

    v16bf ap = load_frag_two(pbuf + ln * SSUB + hi * 8,
                             pbuf + ln * SSUB + 16 + hi * 8);
#pragma unroll
    for (int t = 0; t < 8; ++t) {
      int n0 = (wave * 8 + t) * 16;
      v16bf bf = load_frag_two(xt + (n0 + ln) * SSUB + hi * 8,
                               xt + (n0 + ln) * SSUB + 16 + hi * 8);
      acc[t] = __builtin_amdgcn_wmma_f32_16x16x32_bf16(
          false, ap, false, bf, (short)0, acc[t], false, false);
    }
  }

  // ---- write chunk partials ----
  if (tid < kH) {
    pm[(b * NCHUNK + cchunk) * kH + tid] = mrow[tid];
    pl[(b * NCHUNK + cchunk) * kH + tid] = lrow[tid];
  }
  float* paccb = pacc + ((size_t)(b * NCHUNK + cchunk) * kH) * kD;
#pragma unroll
  for (int t = 0; t < 8; ++t) {
    int n0 = (wave * 8 + t) * 16 + ln;
#pragma unroll
    for (int r = 0; r < 8; ++r) paccb[(r + hi * 8) * kD + n0] = acc[t][r];
  }
}

// =====================================================================
// Kernel C: combine chunk partials -> xa[b,h,:]; ctx = Wv_h @ xa + bv
// =====================================================================
__global__ void __launch_bounds__(256) combine_kernel(
    const float* __restrict__ pm, const float* __restrict__ pl,
    const float* __restrict__ pacc, const float* __restrict__ in_proj_w,
    const float* __restrict__ in_proj_b, float* __restrict__ ctx) {
  __shared__ float xa[kD];
  const int b = blockIdx.x, h = blockIdx.y, tid = threadIdx.x;
  float M = -1e30f;
  for (int c = 0; c < NCHUNK; ++c) M = fmaxf(M, pm[(b * NCHUNK + c) * kH + h]);
  float wgt[NCHUNK];
  float den = 0.f;
  for (int c = 0; c < NCHUNK; ++c) {
    float w = __expf(pm[(b * NCHUNK + c) * kH + h] - M);
    wgt[c] = w;
    den += w * pl[(b * NCHUNK + c) * kH + h];
  }
  float inv = 1.f / den;
  for (int d = tid; d < kD; d += 256) {
    float s = 0.f;
    for (int c = 0; c < NCHUNK; ++c)
      s += wgt[c] * pacc[((size_t)(b * NCHUNK + c) * kH + h) * kD + d];
    xa[d] = s * inv;
  }
  __syncthreads();
  if (tid < 64) {
    const float* wv = in_proj_w + (size_t)(2 * kD + h * 64 + tid) * kD;
    float s = in_proj_b[2 * kD + h * 64 + tid];
    for (int e = 0; e < kD; ++e) s += wv[e] * xa[e];
    ctx[b * kD + h * 64 + tid] = s;
  }
}

// =====================================================================
// Kernel D (one block per batch row): out_proj + LN1 + 16-mode DFT filter
// + proj + LN2 + gate  ->  z3
// =====================================================================
static __device__ float block_sum(float v, float* red) {
  const int tid = threadIdx.x;
  red[tid] = v;
  __syncthreads();
  for (int off = 128; off > 0; off >>= 1) {
    if (tid < off) red[tid] += red[tid + off];
    __syncthreads();
  }
  float r = red[0];
  __syncthreads();
  return r;
}

__global__ void __launch_bounds__(256) post_kernel(
    const float* __restrict__ z_prev, const float* __restrict__ ctx,
    const float* __restrict__ out_w, const float* __restrict__ out_b,
    const float* __restrict__ ln1_g, const float* __restrict__ ln1_b,
    const float* __restrict__ ln2_g, const float* __restrict__ ln2_b,
    const float* __restrict__ scale_real, const float* __restrict__ scale_imag,
    const float* __restrict__ proj_w, const float* __restrict__ proj_b,
    const float* __restrict__ gate, const float* __restrict__ scale_attn,
    const float* __restrict__ scale_denoise, float* __restrict__ z3out) {
  __shared__ float cl[kD], zl[kD], zs[kD];
  __shared__ float red[256];
  __shared__ float cre[16], cim[16];
  __shared__ float zr[16][16], zi[16][16];
  const int b = blockIdx.x, tid = threadIdx.x;
  const float TWO_PI = 6.28318530717958647692f;

  for (int i = tid; i < kD; i += 256) cl[i] = ctx[b * kD + i];
  __syncthreads();

  // attn_out + residual
  float t1v[4];
  float s1 = 0.f, q1 = 0.f;
#pragma unroll
  for (int i = 0; i < 4; ++i) {
    int d = tid + i * 256;
    const float* wr = out_w + (size_t)d * kD;
    float s = out_b[d];
    for (int e = 0; e < kD; ++e) s += wr[e] * cl[e];
    float t = z_prev[b * kD + d] + scale_attn[0] * s;
    t1v[i] = t;
    s1 += t;
    q1 += t * t;
  }
  float mu = block_sum(s1, red) * (1.f / kD);
  float var = block_sum(q1, red) * (1.f / kD) - mu * mu;
  float invs = rsqrtf(var + 1e-5f);
#pragma unroll
  for (int i = 0; i < 4; ++i) {
    int d = tid + i * 256;
    zl[d] = (t1v[i] - mu) * invs * ln1_g[d] + ln1_b[d];
  }
  __syncthreads();

  // filt[m] = mean over D of scale_real/imag
  if (tid < 16) {
    float sr = 0.f, si = 0.f;
    for (int e = 0; e < kD; ++e) { sr += scale_real[tid * kD + e]; si += scale_imag[tid * kD + e]; }
    cre[tid] = sr * (1.f / kD);
    cim[tid] = si * (1.f / kD);
  }
  // DFT partials: Z[m] = sum_n z1[n] e^{-2pi i m n / D}
  {
    int m = tid & 15, seg = tid >> 4;
    float pr = 0.f, pi = 0.f;
    for (int n = seg * 64; n < seg * 64 + 64; ++n) {
      float ang = -TWO_PI * (float)m * (float)n * (1.f / kD);
      float sn, cs;
      __sincosf(ang, &sn, &cs);
      pr += zl[n] * cs;
      pi += zl[n] * sn;
    }
    zr[m][seg] = pr;
    zi[m][seg] = pi;
  }
  __syncthreads();
  if (tid < 16) {
    float Zr = 0.f, Zi = 0.f;
    for (int s = 0; s < 16; ++s) { Zr += zr[tid][s]; Zi += zi[tid][s]; }
    float fr = cre[tid], fi = cim[tid];
    cre[tid] = (fr * Zr - fi * Zi) * (1.f / kD);  // C[m] = filt[m]*Z[m]/D
    cim[tid] = (fr * Zi + fi * Zr) * (1.f / kD);
  }
  __syncthreads();
  // z_spatial[n] = sum_m Re(C[m] e^{+2pi i m n/D})
#pragma unroll
  for (int i = 0; i < 4; ++i) {
    int d = tid + i * 256;
    float s = 0.f;
    for (int m = 0; m < 16; ++m) {
      float ang = TWO_PI * (float)m * (float)d * (1.f / kD);
      float sn, cs;
      __sincosf(ang, &sn, &cs);
      s += cre[m] * cs - cim[m] * sn;
    }
    zs[d] = s;
  }
  __syncthreads();

  // denoised + LN2 + gate
  float t2v[4];
  float s2 = 0.f, q2 = 0.f;
#pragma unroll
  for (int i = 0; i < 4; ++i) {
    int d = tid + i * 256;
    const float* wr = proj_w + (size_t)d * kD;
    float s = proj_b[d];
    for (int e = 0; e < kD; ++e) s += wr[e] * zs[e];
    float t = zl[d] + scale_denoise[0] * s;
    t2v[i] = t;
    s2 += t;
    q2 += t * t;
  }
  float mu2 = block_sum(s2, red) * (1.f / kD);
  float var2 = block_sum(q2, red) * (1.f / kD) - mu2 * mu2;
  float invs2 = rsqrtf(var2 + 1e-5f);
  float g = 1.f / (1.f + __expf(-gate[0]));
#pragma unroll
  for (int i = 0; i < 4; ++i) {
    int d = tid + i * 256;
    float z2 = (t2v[i] - mu2) * invs2 * ln2_g[d] + ln2_b[d];
    z3out[b * kD + d] = g * z2 + (1.f - g) * z_prev[b * kD + d];
  }
}

// =====================================================================
// Kernel E: batch-norm over B=32 per column d
// =====================================================================
__global__ void __launch_bounds__(256) bn_kernel(
    const float* __restrict__ z3, const float* __restrict__ bn_g,
    const float* __restrict__ bn_b, float* __restrict__ out) {
  int d = blockIdx.x * 256 + threadIdx.x;
  float s = 0.f, q = 0.f;
  for (int b = 0; b < kB; ++b) {
    float v = z3[b * kD + d];
    s += v;
    q += v * v;
  }
  float mu = s * (1.f / kB);
  float var = q * (1.f / kB) - mu * mu;
  float inv = rsqrtf(var + 1e-5f);
  float gg = bn_g[d], bb = bn_b[d];
  for (int b = 0; b < kB; ++b)
    out[b * kD + d] = (z3[b * kD + d] - mu) * inv * gg + bb;
}

// =====================================================================
extern "C" void kernel_launch(void* const* d_in, const int* in_sizes, int n_in,
                              void* d_out, int out_size, void* d_ws, size_t ws_size,
                              hipStream_t stream) {
  const float* x_feat    = (const float*)d_in[0];
  const float* z_prev    = (const float*)d_in[1];
  const float* in_proj_w = (const float*)d_in[2];
  const float* in_proj_b = (const float*)d_in[3];
  const float* out_w     = (const float*)d_in[4];
  const float* out_b     = (const float*)d_in[5];
  const float* ln1_g     = (const float*)d_in[6];
  const float* ln1_b     = (const float*)d_in[7];
  const float* ln2_g     = (const float*)d_in[8];
  const float* ln2_b     = (const float*)d_in[9];
  const float* sreal     = (const float*)d_in[10];
  const float* simag     = (const float*)d_in[11];
  const float* proj_w    = (const float*)d_in[12];
  const float* proj_b    = (const float*)d_in[13];
  const float* bn_g      = (const float*)d_in[14];
  const float* bn_b      = (const float*)d_in[15];
  const float* gate      = (const float*)d_in[16];
  const float* s_attn    = (const float*)d_in[17];
  const float* s_den     = (const float*)d_in[18];
  float* out = (float*)d_out;

  // workspace layout (floats)
  float* ws = (float*)d_ws;
  float* qtilde = ws;                                   // 32*16*1024
  float* pm     = qtilde + (size_t)kB * kH * kD;        // 32*8*16
  float* pl     = pm + (size_t)kB * NCHUNK * kH;        // 32*8*16
  float* pacc   = pl + (size_t)kB * NCHUNK * kH;        // 32*8*16*1024
  float* ctx    = pacc + (size_t)kB * NCHUNK * kH * kD; // 32*1024
  float* z3     = ctx + (size_t)kB * kD;                // 32*1024

  const size_t dyn_lds = 65536 + 65536 + 1024 + 2048 + 3 * 16 * 4;  // 134336
  (void)hipFuncSetAttribute((const void*)attn_stream_kernel,
                            hipFuncAttributeMaxDynamicSharedMemorySize,
                            (int)dyn_lds);

  qtilde_kernel<<<dim3(kB, kH), 256, 0, stream>>>(z_prev, in_proj_w, in_proj_b, qtilde);
  attn_stream_kernel<<<dim3(kB, NCHUNK), 256, dyn_lds, stream>>>(x_feat, qtilde, pm, pl, pacc);
  combine_kernel<<<dim3(kB, kH), 256, 0, stream>>>(pm, pl, pacc, in_proj_w, in_proj_b, ctx);
  post_kernel<<<dim3(kB), 256, 0, stream>>>(z_prev, ctx, out_w, out_b, ln1_g, ln1_b,
                                            ln2_g, ln2_b, sreal, simag, proj_w, proj_b,
                                            gate, s_attn, s_den, z3);
  bn_kernel<<<dim3(kD / 256), 256, 0, stream>>>(z3, bn_g, bn_b, out);
}
